// GraphAttentionLayer_86921548136808
// MI455X (gfx1250) — compile-verified, matrix operand order checked by reading, not run on hardware
//
#include <hip/hip_runtime.h>
#include <hip/hip_bf16.h>

#define NN 4096
#define DD 128
#define HH 8
#define KB 64            // keys per attention block-iteration
#define NB (NN / KB)
#define LDK 136          // padded K-tile row pitch (halfs): 272B = 17*16B, breaks bank aliasing
#define LDV 72           // padded V-tile row pitch (halfs): 144B = 9*16B
#define LDP 72           // padded P staging row pitch (halfs)

typedef __attribute__((ext_vector_type(16))) _Float16 v16h;
typedef __attribute__((ext_vector_type(8)))  _Float16 v8h;
typedef __attribute__((ext_vector_type(8)))  float    v8f;

union V16U { v16h v; v8h h[2]; };

// Load a 16x32 f16 tile in the CDNA5 WMMA A-operand layout (also used as the
// B operand when the B matrix is stored transposed):
//   lane L (0..15):  row = row0+L,  halfs 0..7 = K k0+0..7,  8..15 = K k0+16..23
//   lane L+16:       row = row0+L,  halfs 0..7 = K k0+8..15, 8..15 = K k0+24..31
__device__ inline v16h ld_tile(const _Float16* __restrict__ base, int ld, int row0, int k0) {
  int lane = threadIdx.x & 31;
  int hi = lane >> 4, r = lane & 15;
  const _Float16* p = base + (size_t)(row0 + r) * ld + k0 + hi * 8;
  V16U u;
  u.h[0] = *(const v8h*)(p);
  u.h[1] = *(const v8h*)(p + 16);
  return u.v;
}

__device__ inline v8f wmma_f16(v16h a, v16h b, v8f c) {
  return __builtin_amdgcn_wmma_f32_16x16x32_f16(
      /*neg_a=*/false, a, /*neg_b=*/false, b,
      /*c_mod=*/(short)0, c, /*reuse_a=*/false, /*reuse_b=*/false);
}

// Direct global -> LDS 16B async copy (gfx1250, ASYNCcnt-tracked, no VGPR data).
__device__ inline void async_cp16(const _Float16* lds_dst, const _Float16* gsrc) {
  unsigned lds_off = (unsigned)(uintptr_t)lds_dst;  // low 32 bits = LDS byte offset
  asm volatile("global_load_async_to_lds_b128 %0, %1, off"
               :: "v"(lds_off), "v"(gsrc)
               : "memory");
}

__global__ void k_zero(float* __restrict__ p, int n) {
  int i = blockIdx.x * blockDim.x + threadIdx.x;
  if (i < n) p[i] = 0.f;
}

__global__ void k_cvt_x(const float* __restrict__ x, _Float16* __restrict__ xh, int n) {
  int i = blockIdx.x * blockDim.x + threadIdx.x;
  if (i < n) xh[i] = (_Float16)x[i];
}

// Wt layout [3][H][Dout][Din]: Wt[m][h][d][f] = W_m[h][f][d]  (transposed, f16)
__global__ void k_cvt_w(const float* __restrict__ Wk, const float* __restrict__ Wq,
                        const float* __restrict__ Wv, _Float16* __restrict__ Wt) {
  int i = blockIdx.x * blockDim.x + threadIdx.x;
  const int per = HH * DD * DD;
  if (i >= 3 * per) return;
  int m = i / per, rem = i % per;
  int h = rem / (DD * DD); int r2 = rem % (DD * DD);
  int d = r2 / DD, f = r2 % DD;
  const float* src = (m == 0) ? Wk : ((m == 1) ? Wq : Wv);
  Wt[i] = (_Float16)src[(size_t)h * DD * DD + (size_t)f * DD + d];
}

// Projections: per (mat, head) compute [N,128] = xh[N,128] @ W[128,128].
// One wave -> 16 rows x 128 cols.  grid = (N/64, H, 3), block = 128 (4 waves).
// mat 0 -> kh[h][n][d], mat 1 -> qh[h][n][d] (pre-scaled by log2(e) so the
// attention softmax can use raw exp2), mat 2 -> vt[h][d][n] (transposed)
__global__ __launch_bounds__(128) void k_proj(const _Float16* __restrict__ xh,
                                              const _Float16* __restrict__ Wt,
                                              _Float16* __restrict__ qh,
                                              _Float16* __restrict__ kh,
                                              _Float16* __restrict__ vt) {
  int wave = threadIdx.x >> 5;
  int lane = threadIdx.x & 31;
  int row0 = blockIdx.x * 64 + wave * 16;
  int h = blockIdx.y;
  int m = blockIdx.z;
  const _Float16* W = Wt + ((size_t)m * HH + h) * DD * DD;  // [d][f]

  v8f c[8] = {};
  for (int kk = 0; kk < DD; kk += 32) {
    v16h a = ld_tile(xh, DD, row0, kk);
    for (int ct = 0; ct < 8; ct++) {
      v16h b = ld_tile(W, DD, ct * 16, kk);  // B^T rows = output cols
      c[ct] = wmma_f16(a, b, c[ct]);
    }
  }

  int hi = lane >> 4, cl = lane & 15;
  float sc = (m == 1) ? 1.44269504088896f : 1.0f;  // log2(e) folded into Q
  if (m == 2) {
    _Float16* dst = vt + (size_t)h * DD * NN;  // [d][n]
    for (int ct = 0; ct < 8; ct++)
      for (int r = 0; r < 8; r++) {
        int row = row0 + r + 8 * hi;
        int col = ct * 16 + cl;
        dst[(size_t)col * NN + row] = (_Float16)c[ct][r];
      }
  } else {
    _Float16* dst = ((m == 0) ? kh : qh) + (size_t)h * NN * DD;
    for (int ct = 0; ct < 8; ct++)
      for (int r = 0; r < 8; r++) {
        int row = row0 + r + 8 * hi;
        int col = ct * 16 + cl;
        dst[(size_t)row * DD + col] = (_Float16)(c[ct][r] * sc);
      }
  }
}

// Flash attention. One wave owns 16 query rows; block = 4 waves = 64 rows, one head.
// K/V tiles are staged in LDS with double-buffered global_load_async_to_lds_b128.
// grid = (N/64, H).  Accumulates W_m[h] * softmax(QK^T) V into acc[N,D] (f32 atomics).
__global__ __launch_bounds__(128) void k_attn(const _Float16* __restrict__ qh,
                                              const _Float16* __restrict__ kh,
                                              const _Float16* __restrict__ vt,
                                              const float* __restrict__ W_m,
                                              float* __restrict__ acc) {
  __shared__ _Float16 Kb[2][KB][LDK];   // 64 keys x 128 d   (padded)
  __shared__ _Float16 Vb[2][DD][LDV];   // 128 d   x 64 keys (padded)
  __shared__ _Float16 Pb[4][16][LDP];   // per-wave P staging (16 x 64, padded)

  int tid  = threadIdx.x;
  int wave = tid >> 5;
  int lane = tid & 31;
  int h = blockIdx.y;
  int row0 = blockIdx.x * 64 + wave * 16;

  const _Float16* Q = qh + (size_t)h * NN * DD;
  const _Float16* K = kh + (size_t)h * NN * DD;
  const _Float16* V = vt + (size_t)h * DD * NN;  // [d][n]
  float wm = W_m[h];

  // Cooperative async fill of one K/V double-buffer slot: 16 x b128 per thread.
  auto issue = [&](int j, int buf) {
    for (int i = 0; i < 8; i++) {             // K tile: 64 rows x 256B
      int ch = tid + i * 128;
      int kr = ch >> 4, c16 = ch & 15;
      async_cp16(&Kb[buf][kr][c16 * 8], K + (size_t)(j + kr) * DD + c16 * 8);
    }
    for (int i = 0; i < 8; i++) {             // V tile: 128 rows x 128B
      int ch = tid + i * 128;
      int d = ch >> 3, c8 = ch & 7;
      async_cp16(&Vb[buf][d][c8 * 8], V + (size_t)d * NN + j + c8 * 8);
    }
  };

  // Q tiles resident: 16 rows x 128 d, as 4 A-operand tiles.
  v16h qa[4];
  for (int kk = 0; kk < 4; kk++) qa[kk] = ld_tile(Q, DD, row0, kk * 32);

  v8f o[8] = {};                 // O accumulator: 16 rows x 128 cols
  float mrow[8], lrow[8];        // online-softmax state (log2 domain), row = r + 8*hi
  for (int r = 0; r < 8; r++) { mrow[r] = -__builtin_inff(); lrow[r] = 0.f; }

  int hi = lane >> 4, cl = lane & 15;

  issue(0, 0);                   // prologue: prefetch first tile pair

  for (int jb = 0; jb < NB; jb++) {
    int cb = jb & 1;
    if (jb + 1 < NB) {
      issue((jb + 1) * KB, cb ^ 1);                       // prefetch next
      asm volatile("s_wait_asynccnt 0x10" ::: "memory");  // oldest 16 (current) done
    } else {
      asm volatile("s_wait_asynccnt 0x0" ::: "memory");
    }
    __syncthreads();            // current tile visible to all waves

    const _Float16* Kt = &Kb[cb][0][0];
    const _Float16* Vt = &Vb[cb][0][0];

    // S = Q K^T for 64 keys (four 16-key column tiles), f32 accumulate.
    v8f s[4] = {};
    for (int kk = 0; kk < 4; kk++)
      for (int t = 0; t < 4; t++) {
        v16h b = ld_tile(Kt, LDK, t * 16, kk * 32);
        s[t] = wmma_f16(qa[kk], b, s[t]);
      }

    // Online softmax per row (16-lane cross-lane reductions match C layout).
    for (int r = 0; r < 8; r++) {
      float mx = fmaxf(fmaxf(s[0][r], s[1][r]), fmaxf(s[2][r], s[3][r]));
      for (int msk = 1; msk <= 8; msk <<= 1) mx = fmaxf(mx, __shfl_xor(mx, msk, 32));
      float mnew = fmaxf(mrow[r], mx);
      float alpha = exp2f(mrow[r] - mnew);   // Q pre-scaled by log2(e): pure exp2
      float p0 = exp2f(s[0][r] - mnew);
      float p1 = exp2f(s[1][r] - mnew);
      float p2 = exp2f(s[2][r] - mnew);
      float p3 = exp2f(s[3][r] - mnew);
      float rs = (p0 + p1) + (p2 + p3);
      for (int msk = 1; msk <= 8; msk <<= 1) rs += __shfl_xor(rs, msk, 32);
      lrow[r] = lrow[r] * alpha + rs;
      mrow[r] = mnew;
      for (int ct = 0; ct < 8; ct++) o[ct][r] *= alpha;
      int row = r + 8 * hi;                  // C-layout -> A-layout LDS staging
      Pb[wave][row][cl]      = (_Float16)p0;
      Pb[wave][row][16 + cl] = (_Float16)p1;
      Pb[wave][row][32 + cl] = (_Float16)p2;
      Pb[wave][row][48 + cl] = (_Float16)p3;
    }
    asm volatile("s_wait_dscnt 0x0" ::: "memory");  // P writes land before re-read

    v16h pa0 = ld_tile(&Pb[wave][0][0], LDP, 0, 0);   // P keys 0..31
    v16h pa1 = ld_tile(&Pb[wave][0][0], LDP, 0, 32);  // P keys 32..63

    // O += P @ V from the LDS V tile.
    for (int ct = 0; ct < 8; ct++) {
      v16h b0 = ld_tile(Vt, LDV, ct * 16, 0);
      v16h b1 = ld_tile(Vt, LDV, ct * 16, 32);
      o[ct] = wmma_f16(pa0, b0, o[ct]);
      o[ct] = wmma_f16(pa1, b1, o[ct]);
    }
    __syncthreads();            // done reading cb before next prefetch overwrites it
  }

  // Normalize, scale by head mix weight, reduce over heads into acc.
  for (int ct = 0; ct < 8; ct++)
    for (int r = 0; r < 8; r++) {
      int row = row0 + r + 8 * hi;
      int col = ct * 16 + cl;
      float val = o[ct][r] / lrow[r] * wm;
      unsafeAtomicAdd(&acc[(size_t)row * DD + col], val);
    }
}

// out = x + relu((x + acc) @ Wg^T + bg).  One block per row n; tiny FLOPs.
__global__ __launch_bounds__(128) void k_mlp(const float* __restrict__ x,
                                             const float* __restrict__ acc,
                                             const float* __restrict__ Wg,
                                             const float* __restrict__ bg,
                                             float* __restrict__ out) {
  __shared__ float hrow[DD];
  int n = blockIdx.x;
  int d = threadIdx.x;
  float xv = x[(size_t)n * DD + d];
  hrow[d] = xv + acc[(size_t)n * DD + d];
  __syncthreads();
  const float* wr = Wg + (size_t)d * DD;
  float s = bg[d];
  for (int f = 0; f < DD; f++) s += hrow[f] * wr[f];
  out[(size_t)n * DD + d] = xv + fmaxf(s, 0.f);
}

extern "C" void kernel_launch(void* const* d_in, const int* in_sizes, int n_in,
                              void* d_out, int out_size, void* d_ws, size_t ws_size,
                              hipStream_t stream) {
  const float* x  = (const float*)d_in[0];
  const float* Wk = (const float*)d_in[1];
  const float* Wq = (const float*)d_in[2];
  const float* Wv = (const float*)d_in[3];
  const float* Wm = (const float*)d_in[4];
  const float* Wg = (const float*)d_in[5];
  const float* bg = (const float*)d_in[6];
  float* out = (float*)d_out;

  // Workspace carve-up (~28 MB total).
  char* ws = (char*)d_ws;
  size_t off = 0;
  auto carve = [&](size_t bytes) {
    char* p = ws + off;
    off = (off + bytes + 255) & ~(size_t)255;
    return p;
  };
  _Float16* xh  = (_Float16*)carve((size_t)NN * DD * 2);            // x in f16
  _Float16* Wt  = (_Float16*)carve((size_t)3 * HH * DD * DD * 2);   // W^T f16
  _Float16* qh  = (_Float16*)carve((size_t)HH * NN * DD * 2);
  _Float16* kh  = (_Float16*)carve((size_t)HH * NN * DD * 2);
  _Float16* vt  = (_Float16*)carve((size_t)HH * DD * NN * 2);       // V transposed
  float*    acc = (float*)   carve((size_t)NN * DD * 4);            // head-sum f32

  k_zero <<<(NN * DD + 255) / 256, 256, 0, stream>>>(acc, NN * DD);
  k_cvt_x<<<(NN * DD + 255) / 256, 256, 0, stream>>>(x, xh, NN * DD);
  k_cvt_w<<<(3 * HH * DD * DD + 255) / 256, 256, 0, stream>>>(Wk, Wq, Wv, Wt);
  k_proj <<<dim3(NN / 64, HH, 3), 128, 0, stream>>>(xh, Wt, qh, kh, vt);
  k_attn <<<dim3(NN / 64, HH), 128, 0, stream>>>(qh, kh, vt, Wm, acc);
  k_mlp  <<<NN, 128, 0, stream>>>(x, acc, Wg, bg, out);
}
